// DecoderRNN_13700945674915
// MI455X (gfx1250) — matrix operand order repaired and checked by reading, not compiled, not run on hardware
//
#include <hip/hip_runtime.h>
#include <hip/hip_bf16.h>

// ---------------------------------------------------------------------------
// DecoderRNN on MI455X (gfx1250, wave32, WMMA f32<-f16, TDM-fed FC GEMM)
//   B=128 (serial recurrence axis), T=32, E=512, H=1024, V=32000
// ---------------------------------------------------------------------------

#define Bq 128
#define Tq 32
#define Eq 512
#define Hq 1024
#define Vq 32000

typedef _Float16 f16x8   __attribute__((ext_vector_type(8)));
typedef _Float16 f16x16  __attribute__((ext_vector_type(16)));
typedef float    f32x8   __attribute__((ext_vector_type(8)));
typedef unsigned u32x4   __attribute__((ext_vector_type(4)));
typedef int      i32x4v  __attribute__((ext_vector_type(4)));
typedef int      i32x8v  __attribute__((ext_vector_type(8)));

#if defined(__HIP_DEVICE_COMPILE__) && __has_builtin(__builtin_amdgcn_tensor_load_to_lds)
#define USE_TDM 1
#else
#define USE_TDM 0
#endif

// ---- WMMA fragment loaders (wave32 layouts per CDNA5 ISA 7.12.2) ----------
// A (16x32 f16): lanes 0-15 hold row M=lane, K in {0..7,16..23};
//                lanes 16-31 hold row M=lane-16, K in {8..15,24..31}.
__device__ __forceinline__ f16x16 load_a_frag(const _Float16* base, int ld, int lane) {
    int r  = lane & 15;
    int c0 = (lane >> 4) << 3;                 // 0 or 8
    const _Float16* p = base + r * ld;
    f16x8 lo = *reinterpret_cast<const f16x8*>(p + c0);
    f16x8 hi = *reinterpret_cast<const f16x8*>(p + c0 + 16);
    return __builtin_shufflevector(lo, hi, 0,1,2,3,4,5,6,7,8,9,10,11,12,13,14,15);
}

// B (32x16 f16), source stored as rows of W (row j = output column n):
//   lane L: n = L&15, K in {c..c+7, c+16..c+23} with c = (L>>4)*8.
__device__ __forceinline__ f16x16 load_b_frag(const _Float16* base, int ld, int lane) {
    int n = lane & 15;
    int c = (lane >> 4) << 3;                  // 0 or 8
    const _Float16* p = base + n * ld + c;
    f16x8 lo = *reinterpret_cast<const f16x8*>(p);
    f16x8 hi = *reinterpret_cast<const f16x8*>(p + 16);
    return __builtin_shufflevector(lo, hi, 0,1,2,3,4,5,6,7,8,9,10,11,12,13,14,15);
}

__device__ __forceinline__ f32x8 wmma_f16(f16x16 a, f16x16 b, f32x8 c) {
    return __builtin_amdgcn_wmma_f32_16x16x32_f16(false, a, false, b, (short)0, c,
                                                  false, false);
}

__device__ __forceinline__ float sigm(float x) { return 1.0f / (1.0f + __expf(-x)); }

#if USE_TDM
// ---- Tensor Data Mover: 2-D f16 tile (rows x FBKcols) global -> LDS -------
// D# per CDNA5 ISA ch.8: data_size=2B, tile = tile_k x tile_rows, row stride
// `ld` elements; LDS padding 8 dwords every 16 dwords reproduces the
// FLDS=48-element padded row stride used by the WMMA fragment loaders.
__device__ __forceinline__ void tdm_load_tile(unsigned lds_off,
                                              const _Float16* gsrc,
                                              unsigned tile_k, unsigned tile_rows,
                                              unsigned ld, unsigned tensor_rows) {
    unsigned long long ga = (unsigned long long)(size_t)gsrc;
    u32x4 g0;
    g0[0] = 1u;                                      // count=1, user descriptor
    g0[1] = lds_off;                                 // lds_addr (bytes)
    g0[2] = (unsigned)(ga & 0xFFFFFFFFu);            // global_addr[31:0]
    g0[3] = (unsigned)((ga >> 32) & 0x01FFFFFFu)     // global_addr[56:32]
            | (2u << 30);                            // type = 2 (image)
    i32x8v g1;
    g1[0] = (int)((7u << 25)     // pad_amount  = 8 dwords (32B)
                | (3u << 22)     // pad_interval= 16 dwords (64B = one row)
                | (1u << 20)     // pad_enable
                | (1u << 16));   // data_size = 2 bytes; workgroup_mask = 0
    g1[1] = (int)(1024u << 16);                      // tensor_dim0 = 1024 (K)
    g1[2] = (int)((tensor_rows & 0xFFFFu) << 16);    // tensor_dim1 lo16
    g1[3] = (int)(((tensor_rows >> 16) & 0xFFFFu) | (tile_k << 16)); // dim1 hi | tile_dim0
    g1[4] = (int)tile_rows;                          // tile_dim1 (tile_dim2 = 0)
    g1[5] = (int)ld;                                 // tensor_dim0_stride
    g1[6] = 0;
    g1[7] = 0;
    i32x4v z4 = {};
#if __clang_major__ >= 23
    i32x8v z8 = {};
    __builtin_amdgcn_tensor_load_to_lds(g0, g1, z4, z4, z8, 0);
#else
    __builtin_amdgcn_tensor_load_to_lds(g0, g1, z4, z4, 0);
#endif
}
#endif  // USE_TDM

// ---- utility kernels -------------------------------------------------------
__global__ void cvt_f32_to_f16(const float* __restrict__ src,
                               _Float16* __restrict__ dst, int n) {
    for (int i = blockIdx.x * blockDim.x + threadIdx.x; i < n;
         i += gridDim.x * blockDim.x)
        dst[i] = (_Float16)src[i];
}

__global__ void zero_u32(unsigned* __restrict__ p, int n) {
    for (int i = blockIdx.x * blockDim.x + threadIdx.x; i < n;
         i += gridDim.x * blockDim.x)
        p[i] = 0u;
}

// X[b][t][:] = (t==0) ? features[b] : embed_W[captions[b][t]]   (as f16)
__global__ void build_x(const float* __restrict__ features,
                        const int* __restrict__ captions,
                        const float* __restrict__ embed_W,
                        _Float16* __restrict__ X) {
    const int total = Bq * Tq * Eq;
    for (int i = blockIdx.x * blockDim.x + threadIdx.x; i < total;
         i += gridDim.x * blockDim.x) {
        int e = i & (Eq - 1);
        int t = (i >> 9) & (Tq - 1);
        int b = i >> 14;
        float v = (t == 0) ? features[b * Eq + e]
                           : embed_W[(size_t)captions[b * Tq + t] * Eq + e];
        X[i] = (_Float16)v;
    }
}

// ---- fused LSTM step: gates GEMM (WMMA) + cell pointwise -------------------
// grid = H/16 blocks; block = 128 threads = 4 waves; wave g owns gate g.
// Per wave: 2 accumulator tiles (t rows 0-15 and 16-31) x 16 output columns.
__global__ __launch_bounds__(128) void lstm_step(
    const _Float16* __restrict__ xin, int Kx,      // [32, Kx]
    const _Float16* __restrict__ Wx,               // [4H, Kx] row-major
    const _Float16* __restrict__ hprev,            // [32, H]
    const _Float16* __restrict__ Wh,               // [4H, H]
    const float* __restrict__ bi, const float* __restrict__ bh,  // [4H]
    float*     __restrict__ c_state,               // [32, H] in-place
    _Float16*  __restrict__ h_next,                // [32, H]
    _Float16*  __restrict__ ys_out)                // [32, H] or nullptr
{
    const int lane = threadIdx.x & 31;
    const int g    = threadIdx.x >> 5;             // gate index 0..3 (i,f,g,o)
    const int nb   = blockIdx.x << 4;              // 16 output columns

    f32x8 acc0 = {};
    f32x8 acc1 = {};

    // input contribution: x @ Wx^T
    const _Float16* Wr = Wx + (size_t)(g * Hq + nb) * Kx;
    for (int k = 0; k < Kx; k += 32) {
        __builtin_prefetch(Wr + k + 512, 0, 3);    // WGP-scope prefetch
        f16x16 a0 = load_a_frag(xin + k, Kx, lane);
        f16x16 a1 = load_a_frag(xin + 16 * Kx + k, Kx, lane);
        f16x16 bb = load_b_frag(Wr + k, Kx, lane);
        acc0 = wmma_f16(a0, bb, acc0);
        acc1 = wmma_f16(a1, bb, acc1);
    }
    // recurrent contribution: h_prev @ Wh^T
    const _Float16* Wrh = Wh + (size_t)(g * Hq + nb) * Hq;
    for (int k = 0; k < Hq; k += 32) {
        __builtin_prefetch(Wrh + k + 512, 0, 3);
        f16x16 a0 = load_a_frag(hprev + k, Hq, lane);
        f16x16 a1 = load_a_frag(hprev + 16 * Hq + k, Hq, lane);
        f16x16 bb = load_b_frag(Wrh + k, Hq, lane);
        acc0 = wmma_f16(a0, bb, acc0);
        acc1 = wmma_f16(a1, bb, acc1);
    }

    // stage the 4 gate tiles in LDS: gbuf[gate][t][n]
    __shared__ float gbuf[4][32][17];
    {
        const int mrow = (lane >> 4) << 3;   // C/D layout: M = r + 8*(lane>>4)
        const int n    = lane & 15;
#pragma unroll
        for (int r = 0; r < 8; ++r) {
            gbuf[g][mrow + r][n]      = acc0[r];
            gbuf[g][16 + mrow + r][n] = acc1[r];
        }
    }
    __syncthreads();

    // pointwise LSTM cell (torch gate order i,f,g,o), 512 elements
    for (int idx = threadIdx.x; idx < 512; idx += 128) {
        int t  = idx >> 4;
        int nn = idx & 15;
        int gn = nb + nn;
        float xi = gbuf[0][t][nn] + bi[gn]          + bh[gn];
        float xf = gbuf[1][t][nn] + bi[Hq + gn]     + bh[Hq + gn];
        float xg = gbuf[2][t][nn] + bi[2 * Hq + gn] + bh[2 * Hq + gn];
        float xo = gbuf[3][t][nn] + bi[3 * Hq + gn] + bh[3 * Hq + gn];
        float iv = sigm(xi), fv = sigm(xf), gv = tanhf(xg), ov = sigm(xo);
        int   ci = t * Hq + gn;
        float cn = fv * c_state[ci] + iv * gv;
        float hn = ov * tanhf(cn);
        c_state[ci] = cn;
        _Float16 hh = (_Float16)hn;
        h_next[ci] = hh;
        if (ys_out) ys_out[ci] = hh;
    }
}

// ---- final FC: out[4096,32000] = ys[4096,1024] @ fcW^T + fc_b --------------
// 256 threads = 8 waves (2x4); block tile 128x128, K-tile 32.
// Double-buffered LDS fed by the Tensor Data Mover (TENSORcnt-tracked),
// WMMA on one buffer while TDM fills the other.
#define FBM 128
#define FBN 128
#define FBK 32
#define FLDS (FBK + 16)   // padded LDS row stride (96B; matches TDM padding)

__global__ __launch_bounds__(256) void fc_gemm(
    const _Float16* __restrict__ A,    // ys   [4096, 1024]
    const _Float16* __restrict__ Bw,   // fc_W [32000, 1024] (rows = out cols)
    const float* __restrict__ bias,    // [32000]
    float* __restrict__ out)           // [4096, 32000]
{
    __shared__ __align__(32) _Float16 As[2][FBM][FLDS];
    __shared__ __align__(32) _Float16 Bs[2][FBM][FLDS];

    const int tid   = threadIdx.x;
    const int lane  = tid & 31;
    const int wid   = tid >> 5;
    const int waveM = wid & 1;          // 2 waves along M
    const int waveN = wid >> 1;         // 4 waves along N
    const int mblk  = blockIdx.y * FBM;
    const int nblk  = blockIdx.x * FBN;

    f32x8 acc[4][2] = {};

#if USE_TDM
    // prologue: TDM fills buffer 0 (issued once per block, by wave 0)
    if (wid == 0) {
        tdm_load_tile((unsigned)(size_t)&As[0][0][0], A + (size_t)mblk * Hq,
                      FBK, FBM, Hq, Bq * Tq);
        tdm_load_tile((unsigned)(size_t)&Bs[0][0][0], Bw + (size_t)nblk * Hq,
                      FBK, FBM, Hq, Vq);
    }
    int buf = 0;
    for (int kk = 0; kk < Hq; kk += FBK, buf ^= 1) {
        if (wid == 0) __builtin_amdgcn_s_wait_tensorcnt(0);
        __syncthreads();                         // LDS tiles visible to all waves
        if (wid == 0 && (kk + FBK) < Hq) {       // prefetch next K-tile via TDM
            tdm_load_tile((unsigned)(size_t)&As[buf ^ 1][0][0],
                          A + (size_t)mblk * Hq + kk + FBK, FBK, FBM, Hq, Bq * Tq);
            tdm_load_tile((unsigned)(size_t)&Bs[buf ^ 1][0][0],
                          Bw + (size_t)nblk * Hq + kk + FBK, FBK, FBM, Hq, Vq);
        }
        f16x16 bfv[2];
#pragma unroll
        for (int nt = 0; nt < 2; ++nt)
            bfv[nt] = load_b_frag(&Bs[buf][waveN * 32 + nt * 16][0], FLDS, lane);
#pragma unroll
        for (int mt = 0; mt < 4; ++mt) {
            f16x16 af = load_a_frag(&As[buf][waveM * 64 + mt * 16][0], FLDS, lane);
#pragma unroll
            for (int nt = 0; nt < 2; ++nt)
                acc[mt][nt] = wmma_f16(af, bfv[nt], acc[mt][nt]);
        }
        // no trailing barrier needed: next iteration's wait+barrier orders the
        // TDM write into `buf` after every wave finished computing on it.
    }
#else
    // fallback: synchronous vector staging (single buffer)
    const int ldr = tid >> 1;           // 128 rows, 2 threads/row
    const int ldh = (tid & 1) << 4;     // 16-element (32B) halves
    for (int kk = 0; kk < Hq; kk += FBK) {
        *reinterpret_cast<f16x16*>(&As[0][ldr][ldh]) =
            *reinterpret_cast<const f16x16*>(A + (size_t)(mblk + ldr) * Hq + kk + ldh);
        *reinterpret_cast<f16x16*>(&Bs[0][ldr][ldh]) =
            *reinterpret_cast<const f16x16*>(Bw + (size_t)(nblk + ldr) * Hq + kk + ldh);
        __syncthreads();
        f16x16 bfv[2];
#pragma unroll
        for (int nt = 0; nt < 2; ++nt)
            bfv[nt] = load_b_frag(&Bs[0][waveN * 32 + nt * 16][0], FLDS, lane);
#pragma unroll
        for (int mt = 0; mt < 4; ++mt) {
            f16x16 af = load_a_frag(&As[0][waveM * 64 + mt * 16][0], FLDS, lane);
#pragma unroll
            for (int nt = 0; nt < 2; ++nt)
                acc[mt][nt] = wmma_f16(af, bfv[nt], acc[mt][nt]);
        }
        __syncthreads();
    }
#endif

    // epilogue: C/D layout M = r + 8*(lane>>4), N = lane&15
    const int mrow0 = mblk + waveM * 64 + ((lane >> 4) << 3);
    const int ncol0 = nblk + waveN * 32;
#pragma unroll
    for (int mt = 0; mt < 4; ++mt) {
#pragma unroll
        for (int nt = 0; nt < 2; ++nt) {
            int n = ncol0 + nt * 16 + (lane & 15);
            float bv = bias[n];
#pragma unroll
            for (int r = 0; r < 8; ++r) {
                int row = mrow0 + mt * 16 + r;
                out[(size_t)row * Vq + n] = acc[mt][nt][r] + bv;
            }
        }
    }
}

// ---------------------------------------------------------------------------
extern "C" void kernel_launch(void* const* d_in, const int* in_sizes, int n_in,
                              void* d_out, int out_size, void* d_ws, size_t ws_size,
                              hipStream_t stream) {
    const float* features = (const float*)d_in[0];
    const int*   captions = (const int*)d_in[1];
    const float* embed_W  = (const float*)d_in[2];
    const float* W_ih0    = (const float*)d_in[3];
    const float* W_hh0    = (const float*)d_in[4];
    const float* b_ih0    = (const float*)d_in[5];
    const float* b_hh0    = (const float*)d_in[6];
    const float* W_ih1    = (const float*)d_in[7];
    const float* W_hh1    = (const float*)d_in[8];
    const float* b_ih1    = (const float*)d_in[9];
    const float* b_hh1    = (const float*)d_in[10];
    const float* fc_W     = (const float*)d_in[11];
    const float* fc_b     = (const float*)d_in[12];
    float* out = (float*)d_out;

    // workspace layout (all f16 copies of weights live in L2 across launches)
    char*  ws  = (char*)d_ws;
    size_t off = 0;
    auto  carve = [&](size_t bytes) -> void* {
        void* p = ws + off;
        off = (off + bytes + 255) & ~(size_t)255;
        return p;
    };
    _Float16* wih0 = (_Float16*)carve((size_t)4 * Hq * Eq * 2);
    _Float16* whh0 = (_Float16*)carve((size_t)4 * Hq * Hq * 2);
    _Float16* wih1 = (_Float16*)carve((size_t)4 * Hq * Hq * 2);
    _Float16* whh1 = (_Float16*)carve((size_t)4 * Hq * Hq * 2);
    _Float16* wfc  = (_Float16*)carve((size_t)Vq * Hq * 2);
    _Float16* Xf   = (_Float16*)carve((size_t)Bq * Tq * Eq * 2);
    _Float16* ysf  = (_Float16*)carve((size_t)Bq * Tq * Hq * 2);
    _Float16* h0b  = (_Float16*)carve((size_t)2 * Tq * Hq * 2);   // double buffer
    _Float16* h1b  = (_Float16*)carve((size_t)2 * Tq * Hq * 2);
    float*    c0   = (float*)carve((size_t)Tq * Hq * 4);
    float*    c1   = (float*)carve((size_t)Tq * Hq * 4);

    // 1) one-time weight conversions fp32 -> f16
    cvt_f32_to_f16<<<2048, 256, 0, stream>>>(W_ih0, wih0, 4 * Hq * Eq);
    cvt_f32_to_f16<<<2048, 256, 0, stream>>>(W_hh0, whh0, 4 * Hq * Hq);
    cvt_f32_to_f16<<<2048, 256, 0, stream>>>(W_ih1, wih1, 4 * Hq * Hq);
    cvt_f32_to_f16<<<2048, 256, 0, stream>>>(W_hh1, whh1, 4 * Hq * Hq);
    cvt_f32_to_f16<<<4096, 256, 0, stream>>>(fc_W,  wfc,  Vq * Hq);

    // 2) build X (embedding gather + features at t=0)
    build_x<<<2048, 256, 0, stream>>>(features, captions, embed_W, Xf);

    // 3) zero h/c state (h buffers f16 + c buffers f32 are contiguous)
    {
        int words = (2 * Tq * Hq * 2 /*h0*/ + 2 * Tq * Hq * 2 /*h1*/
                     + Tq * Hq * 4 /*c0*/ + Tq * Hq * 4 /*c1*/) / 4;
        zero_u32<<<256, 256, 0, stream>>>((unsigned*)h0b, words);
    }

    // 4) serial recurrence over b = 0..127 (stream order carries the dep)
    for (int b = 0; b < Bq; ++b) {
        int pw = b & 1, pr = pw ^ 1;
        lstm_step<<<Hq / 16, 128, 0, stream>>>(
            Xf + (size_t)b * Tq * Eq, Eq, wih0,
            h0b + (size_t)pr * Tq * Hq, whh0, b_ih0, b_hh0,
            c0, h0b + (size_t)pw * Tq * Hq, (_Float16*)nullptr);
        lstm_step<<<Hq / 16, 128, 0, stream>>>(
            h0b + (size_t)pw * Tq * Hq, Hq, wih1,
            h1b + (size_t)pr * Tq * Hq, whh1, b_ih1, b_hh1,
            c1, h1b + (size_t)pw * Tq * Hq, ysf + (size_t)b * Tq * Hq);
    }

    // 5) vocab projection (compute-bound WMMA GEMM, fc_W f16 resident in L2)
    fc_gemm<<<dim3(Vq / FBN, (Bq * Tq) / FBM), 256, 0, stream>>>(ysf, wfc, fc_b, out);
}
